// TransformerEncoder_85925115724043
// MI455X (gfx1250) — compile-verified, matrix operand order checked by reading, not run on hardware
//
#include <hip/hip_runtime.h>
#include <math.h>

// ---------------------------------------------------------------------------
// Types / constants
// ---------------------------------------------------------------------------
typedef __attribute__((ext_vector_type(16))) _Float16 v16h;
typedef __attribute__((ext_vector_type(8)))  float    v8f;

#define WMMA_F16(a, b, c) \
  __builtin_amdgcn_wmma_f32_16x16x32_f16(false, (a), false, (b), (short)0, (c), false, false)

constexpr int Bsz  = 2;
constexpr int Nseq = 2048;
constexpr int Cdim = 256;
constexpr int Lnum = 4;
constexpr int Hn   = 8;
constexpr int Dh   = 32;          // Cdim / Hn
constexpr int Mrows = Bsz * Nseq; // 4096

// fragment K index for VGPR-pair v (0..7), lane-group offset kb8 in {0,8}
__device__ __forceinline__ int frag_k(int v, int kb8) {
  return (v < 4) ? (kb8 + 2 * v) : (16 + kb8 + 2 * (v - 4));
}

__device__ __forceinline__ float gelu_exact(float x) {
  return 0.5f * x * (1.0f + erff(x * 0.70710678118654752f));
}

// ---------------------------------------------------------------------------
// Positional embedding add: out = x + pe(n, c)
// ---------------------------------------------------------------------------
__global__ __launch_bounds__(256)
void add_pos_kernel(const float* __restrict__ x, float* __restrict__ out) {
  int idx = blockIdx.x * 256 + threadIdx.x;
  int c = idx & (Cdim - 1);
  int n = (idx / Cdim) & (Nseq - 1);
  int i = c >> 1;
  float dv = __expf((float)(2 * i) * (-0.03597789207803197f)); // -ln(1e4)/256
  float ang = (float)n * dv;
  float pe = (c & 1) ? __cosf(ang) : __sinf(ang);
  out[idx] = x[idx] + pe;
}

// ---------------------------------------------------------------------------
// WMMA GEMM: C[M,Nc] = A[M,K] * B[K,Nc] (+bias) (+gelu)
// block = 256 threads (8 waves), block tile 128x64, wave tile 32x32 (4 WMMAs)
// double-buffered LDS: stage tile i+1 while WMMAs consume tile i
// ---------------------------------------------------------------------------
__global__ __launch_bounds__(256)
void gemm_f16wmma(const float* __restrict__ A, const float* __restrict__ Bm,
                  const float* __restrict__ bias, float* __restrict__ Cmat,
                  int M, int K, int Nc, int act) {
  __shared__ _Float16 As[2][128][36];  // [buf][m][k], padded
  __shared__ _Float16 Bs[2][64][36];   // [buf][n][k] (transposed), padded

  const int tid  = threadIdx.x;
  const int lane = tid & 31;
  const int wave = tid >> 5;
  const int m0 = blockIdx.y * 128;
  const int n0 = blockIdx.x * 64;
  const int wm = (wave >> 1) * 32;  // wave row offset (0,32,64,96)
  const int wn = (wave & 1) * 32;   // wave col offset (0,32)
  const int g   = lane >> 4;
  const int l16 = lane & 15;
  const int kb8 = g * 8;

  // staging assignments
  const int arow = tid >> 1;          // 0..127
  const int acol = (tid & 1) * 16;    // 0,16
  const int brow = tid >> 3;          // 0..31
  const int bcol = (tid & 7) * 8;     // 0..56

  auto stage = [&](int buf, int kk) {
    const float4* ap = (const float4*)(A + (size_t)(m0 + arow) * K + kk + acol);
    float4 a0 = ap[0], a1 = ap[1], a2 = ap[2], a3 = ap[3];
    As[buf][arow][acol +  0] = (_Float16)a0.x;  As[buf][arow][acol +  1] = (_Float16)a0.y;
    As[buf][arow][acol +  2] = (_Float16)a0.z;  As[buf][arow][acol +  3] = (_Float16)a0.w;
    As[buf][arow][acol +  4] = (_Float16)a1.x;  As[buf][arow][acol +  5] = (_Float16)a1.y;
    As[buf][arow][acol +  6] = (_Float16)a1.z;  As[buf][arow][acol +  7] = (_Float16)a1.w;
    As[buf][arow][acol +  8] = (_Float16)a2.x;  As[buf][arow][acol +  9] = (_Float16)a2.y;
    As[buf][arow][acol + 10] = (_Float16)a2.z;  As[buf][arow][acol + 11] = (_Float16)a2.w;
    As[buf][arow][acol + 12] = (_Float16)a3.x;  As[buf][arow][acol + 13] = (_Float16)a3.y;
    As[buf][arow][acol + 14] = (_Float16)a3.z;  As[buf][arow][acol + 15] = (_Float16)a3.w;

    const float4* bp = (const float4*)(Bm + (size_t)(kk + brow) * Nc + n0 + bcol);
    float4 b0 = bp[0], b1 = bp[1];
    Bs[buf][bcol + 0][brow] = (_Float16)b0.x;  Bs[buf][bcol + 1][brow] = (_Float16)b0.y;
    Bs[buf][bcol + 2][brow] = (_Float16)b0.z;  Bs[buf][bcol + 3][brow] = (_Float16)b0.w;
    Bs[buf][bcol + 4][brow] = (_Float16)b1.x;  Bs[buf][bcol + 5][brow] = (_Float16)b1.y;
    Bs[buf][bcol + 6][brow] = (_Float16)b1.z;  Bs[buf][bcol + 7][brow] = (_Float16)b1.w;
  };

  v8f acc00 = {}, acc01 = {}, acc10 = {}, acc11 = {};

  stage(0, 0);
  __syncthreads();

  for (int kk = 0; kk < K; kk += 32) {
    const int cur = (kk >> 5) & 1;
    // stage next tile into the other buffer while this one is consumed
    if (kk + 32 < K) {
      stage(cur ^ 1, kk + 32);
      if (kk + 64 < K) {
        __builtin_prefetch(A + (size_t)(m0 + arow) * K + kk + 64 + acol, 0, 0);
        __builtin_prefetch(Bm + (size_t)(kk + 64 + brow) * Nc + n0 + bcol, 0, 0);
      }
    }

    // ---- fragment loads: 2 A-frags, 2 B-frags ----
    v16h af0, af1, bf0, bf1;
#pragma unroll
    for (int v = 0; v < 8; ++v) {
      int kf = frag_k(v, kb8);
      af0[2 * v]     = As[cur][wm + l16][kf];
      af0[2 * v + 1] = As[cur][wm + l16][kf + 1];
      af1[2 * v]     = As[cur][wm + 16 + l16][kf];
      af1[2 * v + 1] = As[cur][wm + 16 + l16][kf + 1];
      bf0[2 * v]     = Bs[cur][wn + l16][kf];
      bf0[2 * v + 1] = Bs[cur][wn + l16][kf + 1];
      bf1[2 * v]     = Bs[cur][wn + 16 + l16][kf];
      bf1[2 * v + 1] = Bs[cur][wn + 16 + l16][kf + 1];
    }
    acc00 = WMMA_F16(af0, bf0, acc00);
    acc01 = WMMA_F16(af0, bf1, acc01);
    acc10 = WMMA_F16(af1, bf0, acc10);
    acc11 = WMMA_F16(af1, bf1, acc11);
    // one barrier per K-step: next-buffer stores done + current reads done
    __syncthreads();
  }

  // ---- epilogue: bias + optional gelu, C/D layout scatter ----
#pragma unroll
  for (int mt = 0; mt < 2; ++mt) {
#pragma unroll
    for (int nt = 0; nt < 2; ++nt) {
      v8f acc = mt ? (nt ? acc11 : acc10) : (nt ? acc01 : acc00);
      int col = n0 + wn + 16 * nt + l16;
      float bv = bias ? bias[col] : 0.0f;
#pragma unroll
      for (int r = 0; r < 8; ++r) {
        int row = m0 + wm + 16 * mt + r + 8 * g;
        float vv = acc[r] + bv;
        if (act) vv = gelu_exact(vv);
        Cmat[(size_t)row * Nc + col] = vv;
      }
    }
  }
}

// ---------------------------------------------------------------------------
// Flash attention: qkv[b, n, {q,k,v}, h, dh] -> out[b, n, h*dh]
// block = 128 threads (4 waves) = 128 queries of one (b,h); each wave owns 32
// queries (2 Q frags). K/V tiles staged in LDS once per block per 32-key step.
// Per step: 4 WMMA (QK^T) + 2 WMMA (rowsum via ones-matrix) + 4 WMMA (PV).
// ---------------------------------------------------------------------------
__global__ __launch_bounds__(128)
void attn_kernel(const float* __restrict__ qkv, float* __restrict__ out) {
  __shared__ _Float16 Ks[32][36];       // [key][dh]            (B-frag for QK^T)
  __shared__ _Float16 Vs[32][36];       // [dh][key] transposed (B-frag for PV)
  __shared__ _Float16 Plds[4][32][36];  // per-wave P scratch (C-layout -> A-layout)

  const int tid  = threadIdx.x;
  const int lane = tid & 31;
  const int wave = tid >> 5;
  const int bh = blockIdx.y;
  const int b  = bh / Hn;
  const int h  = bh % Hn;
  const int q0 = blockIdx.x * 128 + wave * 32;
  const int g   = lane >> 4;
  const int l16 = lane & 15;
  const int kb8 = g * 8;
  const int S3  = 3 * Cdim;  // 768 floats per token
  const float scale = 0.17677669529663687f;  // 1/sqrt(32)
  const size_t baseb = (size_t)b * Nseq * S3;

  // staging assignment: thread t handles token (t/4), dh cols (t%4)*8..+7
  const int skey = tid >> 2;
  const int scol = (tid & 3) * 8;

  // ones B-matrix (32x16 of 1.0) for rowsum-by-WMMA
  v16h onesB;
#pragma unroll
  for (int i = 0; i < 16; ++i) onesB[i] = (_Float16)1.0f;

  // ---- Q fragments (A layout, 16x32 f16 each), loaded once ----
  v16h aQ0, aQ1;
  {
    const float* qp0 = qkv + baseb + (size_t)(q0 + l16) * S3 + h * Dh;
    const float* qp1 = qp0 + (size_t)16 * S3;
#pragma unroll
    for (int v = 0; v < 8; ++v) {
      int kf = frag_k(v, kb8);
      aQ0[2 * v]     = (_Float16)qp0[kf];
      aQ0[2 * v + 1] = (_Float16)qp0[kf + 1];
      aQ1[2 * v]     = (_Float16)qp1[kf];
      aQ1[2 * v + 1] = (_Float16)qp1[kf + 1];
    }
  }

  v8f acc00 = {}, acc01 = {}, acc10 = {}, acc11 = {};
  float mrun[16], lrun[16];
#pragma unroll
  for (int r = 0; r < 16; ++r) { mrun[r] = -INFINITY; lrun[r] = 0.0f; }

  for (int kb = 0; kb < Nseq; kb += 32) {
    // ---- stage K tile (dh-contiguous) and V tile (key-contiguous) ----
    {
      const float* kp = qkv + baseb + (size_t)(kb + skey) * S3 + Cdim + h * Dh + scol;
      const float4* kp4 = (const float4*)kp;
      float4 k0 = kp4[0], k1 = kp4[1];
      Ks[skey][scol + 0] = (_Float16)k0.x;  Ks[skey][scol + 1] = (_Float16)k0.y;
      Ks[skey][scol + 2] = (_Float16)k0.z;  Ks[skey][scol + 3] = (_Float16)k0.w;
      Ks[skey][scol + 4] = (_Float16)k1.x;  Ks[skey][scol + 5] = (_Float16)k1.y;
      Ks[skey][scol + 6] = (_Float16)k1.z;  Ks[skey][scol + 7] = (_Float16)k1.w;

      const float* vp = kp + Cdim;  // V block follows K block
      const float4* vp4 = (const float4*)vp;
      float4 v0 = vp4[0], v1 = vp4[1];
      Vs[scol + 0][skey] = (_Float16)v0.x;  Vs[scol + 1][skey] = (_Float16)v0.y;
      Vs[scol + 2][skey] = (_Float16)v0.z;  Vs[scol + 3][skey] = (_Float16)v0.w;
      Vs[scol + 4][skey] = (_Float16)v1.x;  Vs[scol + 5][skey] = (_Float16)v1.y;
      Vs[scol + 6][skey] = (_Float16)v1.z;  Vs[scol + 7][skey] = (_Float16)v1.w;

      if (kb + 32 < Nseq) {
        __builtin_prefetch(kp + 32 * S3, 0, 0);
        __builtin_prefetch(vp + 32 * S3, 0, 0);
      }
    }
    __syncthreads();

    // ---- K fragments (dh = reduction) + V fragments (key = reduction) ----
    v16h bK0, bK1, bV0, bV1;
#pragma unroll
    for (int v = 0; v < 8; ++v) {
      int kf = frag_k(v, kb8);
      bK0[2 * v]     = Ks[l16][kf];
      bK0[2 * v + 1] = Ks[l16][kf + 1];
      bK1[2 * v]     = Ks[16 + l16][kf];
      bK1[2 * v + 1] = Ks[16 + l16][kf + 1];
      bV0[2 * v]     = Vs[l16][kf];
      bV0[2 * v + 1] = Vs[l16][kf + 1];
      bV1[2 * v]     = Vs[16 + l16][kf];
      bV1[2 * v + 1] = Vs[16 + l16][kf + 1];
    }
    v8f z = {};
    v8f s00 = WMMA_F16(aQ0, bK0, z);
    v8f s01 = WMMA_F16(aQ0, bK1, z);
    v8f s10 = WMMA_F16(aQ1, bK0, z);
    v8f s11 = WMMA_F16(aQ1, bK1, z);

    // ---- online softmax: max via 16-lane shuffles; sum deferred to WMMA ----
#pragma unroll
    for (int f = 0; f < 2; ++f) {
      v8f sa = f ? s10 : s00;
      v8f sb = f ? s11 : s01;
#pragma unroll
      for (int r = 0; r < 8; ++r) {
        int ri = 8 * f + r;
        float a  = sa[r] * scale;
        float bb = sb[r] * scale;
        float t = fmaxf(a, bb);
#pragma unroll
        for (int msk = 1; msk < 16; msk <<= 1) t = fmaxf(t, __shfl_xor(t, msk, 32));
        float mnew  = fmaxf(mrun[ri], t);
        float alpha = __expf(mrun[ri] - mnew);
        float e0 = __expf(a - mnew);
        float e1 = __expf(bb - mnew);
        lrun[ri] *= alpha;
        mrun[ri] = mnew;
        if (f == 0) { acc00[r] *= alpha; acc01[r] *= alpha; }
        else        { acc10[r] *= alpha; acc11[r] *= alpha; }
        Plds[wave][16 * f + r + 8 * g][l16]      = (_Float16)e0;
        Plds[wave][16 * f + r + 8 * g][16 + l16] = (_Float16)e1;
      }
    }
    // barrier: (a) P stores visible for A-frag reload, (b) all waves done
    // reading Ks/Vs before next iteration restages them
    __syncthreads();

    v16h aP0, aP1;
#pragma unroll
    for (int v = 0; v < 8; ++v) {
      int kf = frag_k(v, kb8);
      aP0[2 * v]     = Plds[wave][l16][kf];
      aP0[2 * v + 1] = Plds[wave][l16][kf + 1];
      aP1[2 * v]     = Plds[wave][16 + l16][kf];
      aP1[2 * v + 1] = Plds[wave][16 + l16][kf + 1];
    }
    // rowsum of P via ones-matrix WMMA: every lane gets its row's sum
    v8f sum0 = WMMA_F16(aP0, onesB, z);
    v8f sum1 = WMMA_F16(aP1, onesB, z);
#pragma unroll
    for (int r = 0; r < 8; ++r) {
      lrun[r]     += sum0[r];
      lrun[8 + r] += sum1[r];
    }
    acc00 = WMMA_F16(aP0, bV0, acc00);
    acc01 = WMMA_F16(aP0, bV1, acc01);
    acc10 = WMMA_F16(aP1, bV0, acc10);
    acc11 = WMMA_F16(aP1, bV1, acc11);
  }

  // ---- epilogue: normalize, scatter to [B, N, C] ----
#pragma unroll
  for (int f = 0; f < 2; ++f) {
#pragma unroll
    for (int r = 0; r < 8; ++r) {
      float inv = 1.0f / lrun[8 * f + r];
      int row = q0 + 16 * f + r + 8 * g;
      size_t ob = ((size_t)b * Nseq + row) * Cdim + h * Dh;
      v8f a0 = f ? acc10 : acc00;
      v8f a1 = f ? acc11 : acc01;
      out[ob + l16]      = a0[r] * inv;
      out[ob + 16 + l16] = a1[r] * inv;
    }
  }
}

// ---------------------------------------------------------------------------
// Residual + LayerNorm: out[row] = LN(x[row] + o[row]) * g + b
// one 256-thread block per row (C = 256, one element per thread)
// ---------------------------------------------------------------------------
__global__ __launch_bounds__(256)
void ln_kernel(const float* __restrict__ x, const float* __restrict__ o,
               const float* __restrict__ gam, const float* __restrict__ bet,
               float* __restrict__ out) {
  __shared__ float red[8];
  const int row  = blockIdx.x;
  const int tid  = threadIdx.x;
  const int lane = tid & 31;
  const int wave = tid >> 5;
  const size_t base = (size_t)row * Cdim;

  float v = x[base + tid] + o[base + tid];

  float s = v;
#pragma unroll
  for (int m = 1; m < 32; m <<= 1) s += __shfl_xor(s, m, 32);
  if (lane == 0) red[wave] = s;
  __syncthreads();
  float tot = 0.0f;
#pragma unroll
  for (int w = 0; w < 8; ++w) tot += red[w];
  float mu = tot * (1.0f / Cdim);
  float d = v - mu;
  __syncthreads();

  float s2 = d * d;
#pragma unroll
  for (int m = 1; m < 32; m <<= 1) s2 += __shfl_xor(s2, m, 32);
  if (lane == 0) red[wave] = s2;
  __syncthreads();
  float tot2 = 0.0f;
#pragma unroll
  for (int w = 0; w < 8; ++w) tot2 += red[w];
  float var = tot2 * (1.0f / Cdim);

  out[base + tid] = d * rsqrtf(var + 1e-6f) * gam[tid] + bet[tid];
}

// ---------------------------------------------------------------------------
// Launcher
// ---------------------------------------------------------------------------
extern "C" void kernel_launch(void* const* d_in, const int* in_sizes, int n_in,
                              void* d_out, int out_size, void* d_ws, size_t ws_size,
                              hipStream_t stream) {
  (void)in_sizes; (void)n_in; (void)out_size; (void)ws_size;

  const float* x      = (const float*)d_in[0];
  const float* qkv_w  = (const float*)d_in[1];
  const float* proj_w = (const float*)d_in[2];
  const float* proj_b = (const float*)d_in[3];
  const float* w1     = (const float*)d_in[4];
  const float* b1     = (const float*)d_in[5];
  const float* w2     = (const float*)d_in[6];
  const float* b2     = (const float*)d_in[7];
  const float* g1     = (const float*)d_in[8];
  const float* be1    = (const float*)d_in[9];
  const float* g2     = (const float*)d_in[10];
  const float* be2    = (const float*)d_in[11];
  float* outp = (float*)d_out;

  // workspace partition (floats): xbuf | qkv | t1 | t2  (~24 MB total)
  float* ws   = (float*)d_ws;
  float* xbuf = ws;
  float* qkvb = xbuf + (size_t)Mrows * Cdim;
  float* t1   = qkvb + (size_t)Mrows * 3 * Cdim;
  float* t2   = t1 + (size_t)Mrows * Cdim;

  add_pos_kernel<<<(Mrows * Cdim) / 256, 256, 0, stream>>>(x, xbuf);

  for (int l = 0; l < Lnum; ++l) {
    const float* qw  = qkv_w  + (size_t)l * Cdim * 3 * Cdim;
    const float* pw  = proj_w + (size_t)l * Cdim * Cdim;
    const float* pb  = proj_b + (size_t)l * Cdim;
    const float* w1l = w1  + (size_t)l * Cdim * Cdim;
    const float* b1l = b1  + (size_t)l * Cdim;
    const float* w2l = w2  + (size_t)l * Cdim * Cdim;
    const float* b2l = b2  + (size_t)l * Cdim;
    const float* g1l = g1  + (size_t)l * Cdim;
    const float* e1l = be1 + (size_t)l * Cdim;
    const float* g2l = g2  + (size_t)l * Cdim;
    const float* e2l = be2 + (size_t)l * Cdim;

    // QKV projection: [4096,256] x [256,768]
    gemm_f16wmma<<<dim3((3 * Cdim) / 64, Mrows / 128), 256, 0, stream>>>(
        xbuf, qw, nullptr, qkvb, Mrows, Cdim, 3 * Cdim, 0);

    // attention
    attn_kernel<<<dim3(Nseq / 128, Bsz * Hn), 128, 0, stream>>>(qkvb, t1);

    // output projection
    gemm_f16wmma<<<dim3(Cdim / 64, Mrows / 128), 256, 0, stream>>>(
        t1, pw, pb, t2, Mrows, Cdim, Cdim, 0);

    // x = LN(x + attn_out)
    ln_kernel<<<Mrows, 256, 0, stream>>>(xbuf, t2, g1l, e1l, xbuf);

    // MLP: gelu(x @ w1 + b1) @ w2 + b2
    gemm_f16wmma<<<dim3(Cdim / 64, Mrows / 128), 256, 0, stream>>>(
        xbuf, w1l, b1l, t2, Mrows, Cdim, Cdim, 1);
    gemm_f16wmma<<<dim3(Cdim / 64, Mrows / 128), 256, 0, stream>>>(
        t2, w2l, b2l, t1, Mrows, Cdim, Cdim, 0);

    // x = LN(x + mlp_out); final layer writes straight to d_out
    ln_kernel<<<Mrows, 256, 0, stream>>>(
        xbuf, t1, g2l, e2l, (l == Lnum - 1) ? outp : xbuf);
  }
}
